// ChannelMHAttention_2903397892927
// MI455X (gfx1250) — matrix operand order repaired and checked by reading, not compiled, not run on hardware
//
#include <hip/hip_runtime.h>
#include <hip/hip_bf16.h>

typedef __attribute__((ext_vector_type(16))) _Float16 v16h;
typedef __attribute__((ext_vector_type(8)))  _Float16 v8h;
typedef __attribute__((ext_vector_type(8)))  float    v8f;

#define C_CH   8
#define NQ_S   16
#define NK_S   32
#define BATCH  128
#define D_IN   128
#define E_DIM  512
#define H_HEADS 8
#define A_DIM  64
#define OUT_D  64

__device__ __forceinline__ v16h cat16(v8h lo, v8h hi) {
    v16h r;
#pragma unroll
    for (int j = 0; j < 8; ++j) { r[j] = lo[j]; r[8 + j] = hi[j]; }
    return r;
}

__device__ __forceinline__ v8f wmma_f16(v16h a, v16h b, v8f c) {
    return __builtin_amdgcn_wmma_f32_16x16x32_f16(false, a, false, b, (short)0, c,
                                                  false, false);
}

// ---------------------------------------------------------------------------
// Projection with fused sinusoidal positional encoding:
//   out[cn, m, :] = f16( (A[cn, m, :] + PE(n, :)) @ W[cn, :, :] )
// A: (C*slots, 128, 128) f32   W: (C*slots, 128, 512) f32   out: f16
// grid.x = C*slots*4 (four 128-wide E tiles); block = 256 (8 waves)
// ---------------------------------------------------------------------------
__global__ __launch_bounds__(256) void proj_pe_kernel(
    const float* __restrict__ A, const float* __restrict__ W,
    _Float16* __restrict__ Out, int slots)
{
    const int blk   = blockIdx.x;
    const int ntile = blk & 3;
    const int cn    = blk >> 2;
    const int n     = cn % slots;

    const float* Ab = A + (size_t)cn * BATCH * D_IN;
    const float* Wb = W + (size_t)cn * D_IN * E_DIM + ntile * 128;
    _Float16*    Ob = Out + (size_t)cn * BATCH * E_DIM + ntile * 128;

    __shared__ _Float16 As[128][40] __attribute__((aligned(16)));  // [m][k]
    __shared__ _Float16 Ws[128][40] __attribute__((aligned(16)));  // [nn][k] (transposed)

    const int t    = threadIdx.x;
    const int lane = t & 31;
    const int w    = t >> 5;
    const int half = lane >> 4;
    const int l16  = lane & 15;

    v8f acc[8] = {};

    const float kExp = -9.210340371976184f / (float)D_IN;  // -ln(10000)/D

    for (int k0 = 0; k0 < D_IN; k0 += 32) {
        __syncthreads();
        // A tile 128x32 f32 -> f16 with PE
        {
            const int col4 = (t & 7) * 4;
            const int row0 = t >> 3;  // 0..31
#pragma unroll
            for (int it = 0; it < 4; ++it) {
                const int row = row0 + it * 32;
                const float4 v = *(const float4*)&Ab[row * D_IN + k0 + col4];
                float vals[4] = {v.x, v.y, v.z, v.w};
#pragma unroll
                for (int j = 0; j < 4; ++j) {
                    const int d   = k0 + col4 + j;
                    const float dv  = __expf((float)(d & ~1) * kExp);
                    const float ang = (float)n * dv;
                    const float pe  = (d & 1) ? __cosf(ang) : __sinf(ang);
                    As[row][col4 + j] = (_Float16)(vals[j] + pe);
                }
            }
        }
        // W tile 32x128 f32 -> f16, transposed into Ws[nn][k]
        {
            const int nc4 = (t & 31) * 4;  // 0..124
            const int kr0 = t >> 5;        // 0..7
#pragma unroll
            for (int it = 0; it < 4; ++it) {
                const int kr = kr0 + it * 8;
                const float4 v = *(const float4*)&Wb[(size_t)(k0 + kr) * E_DIM + nc4];
                Ws[nc4 + 0][kr] = (_Float16)v.x;
                Ws[nc4 + 1][kr] = (_Float16)v.y;
                Ws[nc4 + 2][kr] = (_Float16)v.z;
                Ws[nc4 + 3][kr] = (_Float16)v.w;
            }
        }
        __syncthreads();

        const int M = w * 16 + l16;
        const v16h a = cat16(*(const v8h*)&As[M][half * 8],
                             *(const v8h*)&As[M][16 + half * 8]);
#pragma unroll
        for (int nt = 0; nt < 8; ++nt) {
            const int N = nt * 16 + l16;
            const v16h b = cat16(*(const v8h*)&Ws[N][half * 8],
                                 *(const v8h*)&Ws[N][16 + half * 8]);
            acc[nt] = wmma_f16(a, b, acc[nt]);
        }
    }

#pragma unroll
    for (int nt = 0; nt < 8; ++nt)
#pragma unroll
        for (int r = 0; r < 8; ++r) {
            const int m   = w * 16 + half * 8 + r;
            const int col = nt * 16 + l16;
            Ob[(size_t)m * E_DIM + col] = (_Float16)acc[nt][r];
        }
}

// ---------------------------------------------------------------------------
// Attention: one WG per (c,b); wave w == head (heads partition E).
// logits = (Q Kᵀ)/sqrt(64) -> softmax over NK=32 -> attn @ V
// Q:(C,16,B,512) K,V:(C,32,B,512) f16; O:(C,16,B,512) f16
// ---------------------------------------------------------------------------
__global__ __launch_bounds__(256) void attn_kernel(
    const _Float16* __restrict__ Q, const _Float16* __restrict__ K,
    const _Float16* __restrict__ V, _Float16* __restrict__ O)
{
    const int cb = blockIdx.x;
    const int c  = cb >> 7;
    const int b  = cb & 127;

    const int t    = threadIdx.x;
    const int lane = t & 31;
    const int hh   = t >> 5;       // head
    const int half = lane >> 4;
    const int l16  = lane & 15;

    const size_t STRIDE = (size_t)BATCH * E_DIM;  // 65536: slot stride

    __shared__ _Float16 Vt[H_HEADS][64][40]     __attribute__((aligned(16))); // [a][nk]
    __shared__ _Float16 Astage[H_HEADS][16][40] __attribute__((aligned(16))); // attn [q][nk]

    // stage V slice (32 x 64 per head) transposed: lane = nk
    {
        const _Float16* vrow =
            V + ((size_t)c * NK_S * BATCH + b) * E_DIM + (size_t)lane * STRIDE + hh * A_DIM;
#pragma unroll
        for (int a0 = 0; a0 < 64; a0 += 8) {
            const v8h vv = *(const v8h*)&vrow[a0];
#pragma unroll
            for (int j = 0; j < 8; ++j) Vt[hh][a0 + j][lane] = vv[j];
        }
    }
    __syncthreads();

    // ---- logits = Q Kᵀ  (16 x 32, K-loop over A_DIM=64) ----
    const _Float16* qb = Q + ((size_t)c * NQ_S * BATCH + b) * E_DIM + hh * A_DIM;
    const _Float16* kb = K + ((size_t)c * NK_S * BATCH + b) * E_DIM + hh * A_DIM;

    v8f lg[2] = {};
#pragma unroll
    for (int k0 = 0; k0 < A_DIM; k0 += 32) {
        const _Float16* qp = qb + (size_t)l16 * STRIDE + k0 + half * 8;
        const v16h a = cat16(*(const v8h*)qp, *(const v8h*)(qp + 16));
#pragma unroll
        for (int nt = 0; nt < 2; ++nt) {
            const _Float16* kp =
                kb + (size_t)(nt * 16 + l16) * STRIDE + k0 + half * 8;
            const v16h bm = cat16(*(const v8h*)kp, *(const v8h*)(kp + 16));
            lg[nt] = wmma_f16(a, bm, lg[nt]);
        }
    }

    // ---- softmax across the 32 keys (cols live in 16 lanes x 2 regs) ----
    const float inv_temp = 0.125f;  // 1/sqrt(64)
#pragma unroll
    for (int r = 0; r < 8; ++r) {
        float x0 = lg[0][r] * inv_temp;
        float x1 = lg[1][r] * inv_temp;
        float m = fmaxf(x0, x1);
#pragma unroll
        for (int s = 1; s < 16; s <<= 1) m = fmaxf(m, __shfl_xor(m, s, 32));
        const float e0 = __expf(x0 - m);
        const float e1 = __expf(x1 - m);
        float sum = e0 + e1;
#pragma unroll
        for (int s = 1; s < 16; s <<= 1) sum += __shfl_xor(sum, s, 32);
        const float inv = 1.0f / sum;
        Astage[hh][half * 8 + r][l16]      = (_Float16)(e0 * inv);
        Astage[hh][half * 8 + r][16 + l16] = (_Float16)(e1 * inv);
    }
    __syncthreads();

    // ---- out = attn(16x32) @ V(32x64) ----
    const v16h af = cat16(*(const v8h*)&Astage[hh][l16][half * 8],
                          *(const v8h*)&Astage[hh][l16][16 + half * 8]);
    v8f oc[4] = {};
#pragma unroll
    for (int nt = 0; nt < 4; ++nt) {
        const v16h bv = cat16(*(const v8h*)&Vt[hh][nt * 16 + l16][half * 8],
                              *(const v8h*)&Vt[hh][nt * 16 + l16][16 + half * 8]);
        oc[nt] = wmma_f16(af, bv, oc[nt]);
    }

    _Float16* ob = O + ((size_t)c * NQ_S * BATCH + b) * E_DIM + hh * A_DIM;
#pragma unroll
    for (int nt = 0; nt < 4; ++nt)
#pragma unroll
        for (int r = 0; r < 8; ++r) {
            const int q = half * 8 + r;
            const int a = nt * 16 + l16;
            ob[(size_t)q * STRIDE + a] = (_Float16)oc[nt][r];
        }
}

// ---------------------------------------------------------------------------
// Output projection per (c,q): (128 x 512 f16) @ (512 x 64 f32) -> f32
// grid.x = C*NQ; block = 256 (8 waves, each a 16-row strip)
// ---------------------------------------------------------------------------
__global__ __launch_bounds__(256) void outproj_kernel(
    const _Float16* __restrict__ A, const float* __restrict__ Wp,
    float* __restrict__ Out)
{
    const int cq = blockIdx.x;
    const _Float16* Ab = A + (size_t)cq * BATCH * E_DIM;
    const float*    Wb = Wp + (size_t)cq * E_DIM * OUT_D;
    float*          Ob = Out + (size_t)cq * BATCH * OUT_D;

    __shared__ _Float16 Wt[64][40] __attribute__((aligned(16)));  // [o][k]

    const int t    = threadIdx.x;
    const int lane = t & 31;
    const int w    = t >> 5;
    const int half = lane >> 4;
    const int l16  = lane & 15;

    v8f acc[4] = {};

    for (int k0 = 0; k0 < E_DIM; k0 += 32) {
        __syncthreads();
        {   // stage Wp tile 32x64 f32 -> f16 transposed
            const int o4  = (t & 15) * 4;  // 0..60
            const int kr0 = t >> 4;        // 0..15
#pragma unroll
            for (int it = 0; it < 2; ++it) {
                const int kr = kr0 + it * 16;
                const float4 v = *(const float4*)&Wb[(size_t)(k0 + kr) * OUT_D + o4];
                Wt[o4 + 0][kr] = (_Float16)v.x;
                Wt[o4 + 1][kr] = (_Float16)v.y;
                Wt[o4 + 2][kr] = (_Float16)v.z;
                Wt[o4 + 3][kr] = (_Float16)v.w;
            }
        }
        __syncthreads();

        const int M = w * 16 + l16;
        const _Float16* ap = Ab + (size_t)M * E_DIM + k0 + half * 8;
        const v16h a = cat16(*(const v8h*)ap, *(const v8h*)(ap + 16));
#pragma unroll
        for (int nt = 0; nt < 4; ++nt) {
            const v16h bv = cat16(*(const v8h*)&Wt[nt * 16 + l16][half * 8],
                                  *(const v8h*)&Wt[nt * 16 + l16][16 + half * 8]);
            acc[nt] = wmma_f16(a, bv, acc[nt]);
        }
    }

#pragma unroll
    for (int nt = 0; nt < 4; ++nt)
#pragma unroll
        for (int r = 0; r < 8; ++r) {
            const int m = w * 16 + half * 8 + r;
            Ob[(size_t)m * OUT_D + nt * 16 + l16] = acc[nt][r];
        }
}

extern "C" void kernel_launch(void* const* d_in, const int* in_sizes, int n_in,
                              void* d_out, int out_size, void* d_ws, size_t ws_size,
                              hipStream_t stream) {
    (void)in_sizes; (void)n_in; (void)out_size; (void)ws_size;
    const float* q  = (const float*)d_in[0];
    const float* k  = (const float*)d_in[1];
    const float* Wq = (const float*)d_in[2];
    const float* Wk = (const float*)d_in[3];
    const float* Wv = (const float*)d_in[4];
    const float* Wp = (const float*)d_in[5];

    const size_t QSZ = (size_t)C_CH * NQ_S * BATCH * E_DIM;  // 8.4M f16
    const size_t KSZ = (size_t)C_CH * NK_S * BATCH * E_DIM;  // 16.8M f16

    _Float16* qf = (_Float16*)d_ws;
    _Float16* kf = qf + QSZ;
    _Float16* vf = kf + KSZ;
    _Float16* of = vf + KSZ;

    proj_pe_kernel<<<C_CH * NQ_S * 4, 256, 0, stream>>>(q, Wq, qf, NQ_S);
    proj_pe_kernel<<<C_CH * NK_S * 4, 256, 0, stream>>>(k, Wk, kf, NK_S);
    proj_pe_kernel<<<C_CH * NK_S * 4, 256, 0, stream>>>(k, Wv, vf, NK_S);
    attn_kernel<<<C_CH * BATCH, 256, 0, stream>>>(qf, kf, vf, of);
    outproj_kernel<<<C_CH * NQ_S, 256, 0, stream>>>(of, Wp, (float*)d_out);
}
